// LLaMACausalSelfAttention_10161892622750
// MI455X (gfx1250) — matrix operand order repaired and checked by reading, not compiled, not run on hardware
//
#include <hip/hip_runtime.h>
#include <cstdint>

// ---------------------------------------------------------------------------
// LLaMA causal self-attention for MI455X (gfx1250, wave32, WMMA bf16).
// B=1, S=2048, DIM=2048, H=32, KVH=8, D=64.
// ---------------------------------------------------------------------------

#define DEV __device__ __forceinline__

typedef __attribute__((ext_vector_type(4)))  unsigned int u32x4;
typedef __attribute__((ext_vector_type(4)))  int          i32x4;
typedef __attribute__((ext_vector_type(8)))  float        v8f;
typedef __attribute__((ext_vector_type(16))) __bf16       v16bf;

union Frag { v16bf v; u32x4 q[2]; };

constexpr int S_   = 2048;
constexpr int DIM_ = 2048;
constexpr int H_   = 32;
constexpr int KVH_ = 8;
constexpr int D_   = 64;

DEV unsigned short f2bf(float x) {
  unsigned int u = __float_as_uint(x);
  u += 0x7FFFu + ((u >> 16) & 1u);       // round-to-nearest-even
  return (unsigned short)(u >> 16);
}

DEV unsigned int f2bf_pack(float lo, float hi) {
  return (unsigned int)f2bf(lo) | ((unsigned int)f2bf(hi) << 16);
}

DEV v8f wmma_bf16(const Frag& a, const Frag& b, v8f c) {
  // v_wmma_f32_16x16x32_bf16: D = A(16x32 bf16) * B(32x16 bf16) + C(16x16 f32)
  return __builtin_amdgcn_wmma_f32_16x16x32_bf16(
      /*neg_a=*/false, a.v, /*neg_b=*/false, b.v,
      /*c_mod=*/(short)0, c, /*reuse_a=*/false, /*reuse_b=*/false);
}

// ---------------------------------------------------------------------------
// CDNA5 async global->LDS copy (16B per lane), tracked by ASYNCcnt.
// Builtin convention (as llvm.amdgcn.global.load.lds): (gsrc, lds_dst, off, cpol)
// ---------------------------------------------------------------------------
DEV void async_cp16(void* lds_dst, const void* gsrc) {
#if __has_builtin(__builtin_amdgcn_global_load_async_to_lds_b128)
  __builtin_amdgcn_global_load_async_to_lds_b128(
      (i32x4*)(void*)(uintptr_t)gsrc, (i32x4*)lds_dst, 0, 0);
#else
  asm volatile("global_load_async_to_lds_b128 %0, %1, off"
               :: "v"((unsigned int)(uintptr_t)lds_dst),
                  "v"((unsigned long long)(uintptr_t)gsrc)
               : "memory");
#endif
}

DEV void wait_async4() {
#if __has_builtin(__builtin_amdgcn_s_wait_asynccnt)
  __builtin_amdgcn_s_wait_asynccnt(4);
#else
  asm volatile("s_wait_asynccnt 0x4" ::: "memory");
#endif
}

DEV void wait_async0() {
#if __has_builtin(__builtin_amdgcn_s_wait_asynccnt)
  __builtin_amdgcn_s_wait_asynccnt(0);
#else
  asm volatile("s_wait_asynccnt 0x0" ::: "memory");
#endif
}

// ---------------------------------------------------------------------------
// Pack fp32 row-major A [M][K] into WMMA A-fragment tiles (16x32, bf16).
//   lane<16 : M=lane,    K = 0..15
//   lane>=16: M=lane-16, K = 16..31
// Tile = 32 lanes x 16 bf16 contiguous (1KB).
// ---------------------------------------------------------------------------
__global__ void __launch_bounds__(256) pack_a_kernel(const float* __restrict__ src,
                                                     unsigned short* __restrict__ dst,
                                                     int M, int K) {
  int KT = K >> 5;
  int t = blockIdx.x * blockDim.x + threadIdx.x;       // one thread per (tile,lane)
  int total = (M >> 4) * KT * 32;
  if (t >= total) return;
  int lane = t & 31;
  int tile = t >> 5;
  int kt = tile % KT;
  int mt = tile / KT;
  int row = mt * 16 + (lane & 15);
  const float* s = src + (size_t)row * K + (size_t)kt * 32;
  unsigned short* d = dst + (size_t)tile * 512 + lane * 16;
  int kb = (lane < 16) ? 0 : 16;
  unsigned int wbuf[8];
#pragma unroll
  for (int p = 0; p < 8; ++p) {
    int i0 = 2 * p, i1 = 2 * p + 1;
    int k0 = kb + i0;
    int k1 = kb + i1;
    wbuf[p] = f2bf_pack(s[k0], s[k1]);
  }
  u32x4* dv = (u32x4*)d;
  dv[0] = u32x4{wbuf[0], wbuf[1], wbuf[2], wbuf[3]};
  dv[1] = u32x4{wbuf[4], wbuf[5], wbuf[6], wbuf[7]};
}

// ---------------------------------------------------------------------------
// Pack fp32 row-major B [K][N] into WMMA B-fragment tiles (32x16, bf16).
//   lane<16 : N=lane,    K = 0..15 ;  lane>=16: N=lane-16, K = 16..31
// ---------------------------------------------------------------------------
__global__ void __launch_bounds__(256) pack_b_kernel(const float* __restrict__ src,
                                                     unsigned short* __restrict__ dst,
                                                     int K, int N) {
  int KT = K >> 5;
  int t = blockIdx.x * blockDim.x + threadIdx.x;
  int total = (N >> 4) * KT * 32;
  if (t >= total) return;
  int lane = t & 31;
  int tile = t >> 5;
  int kt = tile % KT;
  int nt = tile / KT;
  int col = nt * 16 + (lane & 15);
  int krow0 = kt * 32 + ((lane < 16) ? 0 : 16);
  const float* s = src + (size_t)krow0 * N + col;
  unsigned short* d = dst + (size_t)tile * 512 + lane * 16;
  unsigned int wbuf[8];
#pragma unroll
  for (int p = 0; p < 8; ++p)
    wbuf[p] = f2bf_pack(s[(size_t)(2 * p) * N], s[(size_t)(2 * p + 1) * N]);
  u32x4* dv = (u32x4*)d;
  dv[0] = u32x4{wbuf[0], wbuf[1], wbuf[2], wbuf[3]};
  dv[1] = u32x4{wbuf[4], wbuf[5], wbuf[6], wbuf[7]};
}

// ---------------------------------------------------------------------------
// GEMM: C[M][N] f32 = Apacked * Bpacked + bias.
// Block tile 128x128 (8 waves as 4Mx2N, wave tile 32x64). K-panel double-
// buffered in LDS, filled by async global->LDS b128 copies (ASYNCcnt).
// Per K-step: 16KB global -> 64 WMMAs/block (~64 FLOP per global byte).
// ---------------------------------------------------------------------------
__global__ void __launch_bounds__(256) gemm_bf16_kernel(
    const unsigned short* __restrict__ Ap, const unsigned short* __restrict__ Bp,
    const float* __restrict__ bias, float* __restrict__ C, int M, int N, int K) {
  __shared__ __align__(16) unsigned short smA[2][8][512];   // 8 A tiles / buffer
  __shared__ __align__(16) unsigned short smB[2][8][512];   // 8 B tiles / buffer

  int KT = K >> 5;
  int lane = threadIdx.x & 31;
  int w = threadIdx.x >> 5;
  int l15 = lane & 15;
  int mt0 = blockIdx.x * 8;            // 8 M-tiles per block
  int nt0 = blockIdx.y * 8;            // 8 N-tiles per block
  int wm = w >> 1;                     // 0..3
  int wn = w & 1;                      // 0..1

  auto issue = [&](int kt, int bsel) {
    // wave w stages A tile (mt0+w, kt) and B tile (nt0+w, kt): 4 async b128 ops
    const unsigned short* ga = Ap + ((size_t)(mt0 + w) * KT + kt) * 512 + lane * 16;
    const unsigned short* gb = Bp + ((size_t)(nt0 + w) * KT + kt) * 512 + lane * 16;
    async_cp16(&smA[bsel][w][lane * 16],     ga);
    async_cp16(&smA[bsel][w][lane * 16 + 8], ga + 8);
    async_cp16(&smB[bsel][w][lane * 16],     gb);
    async_cp16(&smB[bsel][w][lane * 16 + 8], gb + 8);
  };

  const v8f Z = {0.f, 0.f, 0.f, 0.f, 0.f, 0.f, 0.f, 0.f};
  v8f acc[2][4] = {{Z, Z, Z, Z}, {Z, Z, Z, Z}};

  issue(0, 0);
  for (int kt = 0; kt < KT; ++kt) {
    int bsel = kt & 1;
    if (kt + 1 < KT) { issue(kt + 1, bsel ^ 1); wait_async4(); }
    else             { wait_async0(); }
    __syncthreads();                   // buffer bsel now valid for all waves

    Frag a0, a1;
    const unsigned short* pa0 = &smA[bsel][2 * wm][lane * 16];
    const unsigned short* pa1 = &smA[bsel][2 * wm + 1][lane * 16];
    a0.q[0] = *(const u32x4*)(pa0); a0.q[1] = *(const u32x4*)(pa0 + 8);
    a1.q[0] = *(const u32x4*)(pa1); a1.q[1] = *(const u32x4*)(pa1 + 8);
#pragma unroll
    for (int j = 0; j < 4; ++j) {
      const unsigned short* pb = &smB[bsel][4 * wn + j][lane * 16];
      Frag b;
      b.q[0] = *(const u32x4*)(pb);
      b.q[1] = *(const u32x4*)(pb + 8);
      acc[0][j] = wmma_bf16(a0, b, acc[0][j]);
      acc[1][j] = wmma_bf16(a1, b, acc[1][j]);
    }
    __syncthreads();                   // reads done before buffer is re-filled
  }

  int rowoff = (lane < 16) ? 0 : 8;
#pragma unroll
  for (int i = 0; i < 2; ++i) {
    int mbase = (mt0 + 2 * wm + i) * 16;
#pragma unroll
    for (int j = 0; j < 4; ++j) {
      int col = (nt0 + 4 * wn + j) * 16 + l15;
      float bb = bias ? bias[col] : 0.f;
#pragma unroll
      for (int r = 0; r < 8; ++r)
        C[(size_t)(mbase + r + rowoff) * N + col] = acc[i][j][r] + bb;
    }
  }
}

// ---------------------------------------------------------------------------
// RoPE + bf16 repack for Q: qraw f32 [S][2048] -> Qb bf16 [H][S][64]
// ---------------------------------------------------------------------------
__global__ void __launch_bounds__(256) rope_pack_q_kernel(
    const float* __restrict__ qraw, const float* __restrict__ cosb,
    const float* __restrict__ sinb, unsigned short* __restrict__ Qb) {
  int idx = blockIdx.x * blockDim.x + threadIdx.x;
  if (idx >= S_ * (DIM_ / 2)) return;
  int s = idx >> 10;
  int p = idx & 1023;
  int col = p * 2;
  int h = col >> 6;
  int d = col & 63;
  float t0 = qraw[(size_t)s * DIM_ + col];
  float t1 = qraw[(size_t)s * DIM_ + col + 1];
  float c  = cosb[s * 32 + (d >> 1)];
  float sn = sinb[s * 32 + (d >> 1)];
  size_t base = ((size_t)h * S_ + s) * 64 + d;
  *(unsigned int*)(Qb + base) = f2bf_pack(t0 * c - t1 * sn, t0 * sn + t1 * c);
}

// RoPE + bf16 repack for K: kraw f32 [S][512] -> Kb bf16 [KVH][S][64]
__global__ void __launch_bounds__(256) rope_pack_k_kernel(
    const float* __restrict__ kraw, const float* __restrict__ cosb,
    const float* __restrict__ sinb, unsigned short* __restrict__ Kb) {
  int idx = blockIdx.x * blockDim.x + threadIdx.x;
  if (idx >= S_ * 256) return;
  int s = idx >> 8;
  int p = idx & 255;
  int col = p * 2;
  int kvh = col >> 6;
  int d = col & 63;
  float t0 = kraw[(size_t)s * 512 + col];
  float t1 = kraw[(size_t)s * 512 + col + 1];
  float c  = cosb[s * 32 + (d >> 1)];
  float sn = sinb[s * 32 + (d >> 1)];
  size_t base = ((size_t)kvh * S_ + s) * 64 + d;
  *(unsigned int*)(Kb + base) = f2bf_pack(t0 * c - t1 * sn, t0 * sn + t1 * c);
}

// V transpose + bf16: vraw f32 [S][512] -> Vt bf16 [KVH][64][S]
__global__ void __launch_bounds__(256) pack_v_kernel(const float* __restrict__ vraw,
                                                     unsigned short* __restrict__ Vt) {
  int idx = blockIdx.x * blockDim.x + threadIdx.x;
  if (idx >= S_ * 512) return;
  int s = idx >> 9;
  int col = idx & 511;
  int kvh = col >> 6;
  int d = col & 63;
  Vt[((size_t)(kvh * 64 + d)) * S_ + s] = f2bf(vraw[(size_t)s * 512 + col]);
}

// ---------------------------------------------------------------------------
// Flash attention: one wave per (head, 16-row query tile). Online softmax in
// f32, probabilities staged bf16 through LDS (C-frag -> A-frag transpose),
// both scores and P*V on v_wmma_f32_16x16x32_bf16.
// ---------------------------------------------------------------------------
__global__ void __launch_bounds__(256) attn_kernel(
    const unsigned short* __restrict__ Qb,   // [H][S][64] bf16
    const unsigned short* __restrict__ Kb,   // [KVH][S][64] bf16
    const unsigned short* __restrict__ Vt,   // [KVH][64][S] bf16
    float* __restrict__ Oout) {              // [S][H*64] f32
  __shared__ __align__(16) unsigned short pst[8][16][32];  // per-wave P staging

  int lane = threadIdx.x & 31;
  int w = threadIdx.x >> 5;
  int gid = blockIdx.x * 8 + w;      // (head, qtile)
  int h = gid >> 7;                  // / (S/16 = 128)
  int qt = gid & 127;
  int kvh = h >> 2;                  // GQA: 4 query heads per kv head
  int q0 = qt * 16;
  int l15 = lane & 15;
  int hi = lane >> 4;                // half-wave select
  int rowoff = hi * 8;
  const float scale = 0.125f;        // 1/sqrt(64)

  // Preload Q A-fragments (d-chunks c=0,1), fixed for this wave.
  Frag qf[2];
  {
    const unsigned short* qrow = Qb + ((size_t)(h * S_ + q0 + l15)) * 64;
#pragma unroll
    for (int c = 0; c < 2; ++c) {
      int o1 = c * 32 + (hi ? 16 : 0);
      int o2 = c * 32 + (hi ? 24 : 8);
      qf[c].q[0] = *(const u32x4*)(qrow + o1);
      qf[c].q[1] = *(const u32x4*)(qrow + o2);
    }
  }

  const v8f Z = {0.f, 0.f, 0.f, 0.f, 0.f, 0.f, 0.f, 0.f};
  v8f accO[4] = {Z, Z, Z, Z};
  float mrow[8], lrow[8];
#pragma unroll
  for (int r = 0; r < 8; ++r) { mrow[r] = -1e30f; lrow[r] = 0.f; }

  int nch = ((q0 + 15) >> 5) + 1;    // causal: chunks of 32 keys
  for (int kc = 0; kc < nch; ++kc) {
    int k0 = kc * 32;

    // ---- scores: two 16x16 tiles (keys k0..+15 and k0+16..+31) ----
    v8f sc0 = Z, sc1 = Z;
    {
      const unsigned short* kr0 =
          Kb + ((size_t)(kvh * S_ + k0 + l15)) * 64 + (hi ? 16 : 0);
      const unsigned short* kr1 =
          Kb + ((size_t)(kvh * S_ + k0 + 16 + l15)) * 64 + (hi ? 16 : 0);
#pragma unroll
      for (int c = 0; c < 2; ++c) {
        Frag b0, b1;
        b0.q[0] = *(const u32x4*)(kr0 + c * 32);
        b0.q[1] = *(const u32x4*)(kr0 + c * 32 + 8);
        b1.q[0] = *(const u32x4*)(kr1 + c * 32);
        b1.q[1] = *(const u32x4*)(kr1 + c * 32 + 8);
        sc0 = wmma_bf16(qf[c], b0, sc0);
        sc1 = wmma_bf16(qf[c], b1, sc1);
      }
    }

    // ---- causal mask + online softmax over the 32-key chunk ----
#pragma unroll
    for (int r = 0; r < 8; ++r) {
      int qrow = q0 + r + rowoff;
      float s0 = sc0[r] * scale; if (k0 + l15 > qrow)      s0 = -1e30f;
      float s1 = sc1[r] * scale; if (k0 + 16 + l15 > qrow) s1 = -1e30f;
      float t = fmaxf(s0, s1);
      t = fmaxf(t, __shfl_xor(t, 1, 32));
      t = fmaxf(t, __shfl_xor(t, 2, 32));
      t = fmaxf(t, __shfl_xor(t, 4, 32));
      t = fmaxf(t, __shfl_xor(t, 8, 32));
      float mnew = fmaxf(mrow[r], t);
      float p0 = __expf(s0 - mnew);
      float p1 = __expf(s1 - mnew);
      float su = p0 + p1;
      su += __shfl_xor(su, 1, 32);
      su += __shfl_xor(su, 2, 32);
      su += __shfl_xor(su, 4, 32);
      su += __shfl_xor(su, 8, 32);
      float f = __expf(mrow[r] - mnew);
      lrow[r] = lrow[r] * f + su;
      mrow[r] = mnew;
#pragma unroll
      for (int j = 0; j < 4; ++j) accO[j][r] = accO[j][r] * f;
      pst[w][r + rowoff][l15]      = f2bf(p0);
      pst[w][r + rowoff][16 + l15] = f2bf(p1);
    }

    asm volatile("s_wait_dscnt 0" ::: "memory");   // LDS RAW within the wave

    // ---- P A-fragment (16x32) from LDS, then P*V over d = 0..63 ----
    Frag pf;
    {
      const unsigned short* pr = &pst[w][l15][0];
      if (hi) { pf.q[0] = *(const u32x4*)(pr + 16); pf.q[1] = *(const u32x4*)(pr + 24); }
      else    { pf.q[0] = *(const u32x4*)(pr);      pf.q[1] = *(const u32x4*)(pr + 8); }
    }
#pragma unroll
    for (int j = 0; j < 4; ++j) {
      const unsigned short* vr =
          Vt + ((size_t)(kvh * 64 + j * 16 + l15)) * S_ + k0 + (hi ? 16 : 0);
      Frag b;
      b.q[0] = *(const u32x4*)(vr);
      b.q[1] = *(const u32x4*)(vr + 8);
      accO[j] = wmma_bf16(pf, b, accO[j]);
    }
  }

  // ---- normalize and write f32 attention output [s][h*64+d] ----
#pragma unroll
  for (int j = 0; j < 4; ++j) {
    int col = h * 64 + j * 16 + l15;
#pragma unroll
    for (int r = 0; r < 8; ++r) {
      float inv = 1.0f / lrow[r];
      Oout[(size_t)(q0 + r + rowoff) * DIM_ + col] = accO[j][r] * inv;
    }
  }
}

// ---------------------------------------------------------------------------
// Host-side orchestration
// ---------------------------------------------------------------------------
extern "C" void kernel_launch(void* const* d_in, const int* in_sizes, int n_in,
                              void* d_out, int out_size, void* d_ws, size_t ws_size,
                              hipStream_t stream) {
  const float* x    = (const float*)d_in[0];
  const float* fcos = (const float*)d_in[1];
  const float* fsin = (const float*)d_in[2];
  // d_in[3] mask: causal handled analytically
  const float* Wq = (const float*)d_in[4];
  const float* bq = (const float*)d_in[5];
  const float* Wk = (const float*)d_in[6];
  const float* bk = (const float*)d_in[7];
  const float* Wv = (const float*)d_in[8];
  const float* bv = (const float*)d_in[9];
  const float* Wo = (const float*)d_in[10];
  const float* bo = (const float*)d_in[11];
  float* out = (float*)d_out;

  char* ws = (char*)d_ws;
  size_t off = 0;
  auto take = [&](size_t bytes) -> char* {
    char* p = ws + off;
    off += (bytes + 255) & ~(size_t)255;
    return p;
  };
  unsigned short* Xp    = (unsigned short*)take((size_t)S_ * DIM_ * 2);      // x packed-A
  unsigned short* Wqp   = (unsigned short*)take((size_t)DIM_ * DIM_ * 2);
  unsigned short* Wkp   = (unsigned short*)take((size_t)DIM_ * 512 * 2);
  unsigned short* Wvp   = (unsigned short*)take((size_t)DIM_ * 512 * 2);
  unsigned short* Wop   = (unsigned short*)take((size_t)DIM_ * DIM_ * 2);
  float*          qraw  = (float*)take((size_t)S_ * DIM_ * 4);
  float*          kraw  = (float*)take((size_t)S_ * 512 * 4);
  float*          vraw  = (float*)take((size_t)S_ * 512 * 4);
  unsigned short* Qb    = (unsigned short*)take((size_t)H_ * S_ * 64 * 2);
  unsigned short* Kb    = (unsigned short*)take((size_t)KVH_ * S_ * 64 * 2);
  unsigned short* Vt    = (unsigned short*)take((size_t)KVH_ * 64 * S_ * 2);
  float*          attnO = (float*)take((size_t)S_ * DIM_ * 4);
  unsigned short* Ap2   = (unsigned short*)take((size_t)S_ * DIM_ * 2);      // attn out packed-A
  (void)ws_size; (void)in_sizes; (void)n_in; (void)out_size;

  // 1) pack activations and weights to bf16 WMMA fragment layout
  {
    int tot = (S_ / 16) * (DIM_ / 32) * 32;
    pack_a_kernel<<<(tot + 255) / 256, 256, 0, stream>>>(x, Xp, S_, DIM_);
  }
  {
    int tot = (DIM_ / 16) * (DIM_ / 32) * 32;
    pack_b_kernel<<<(tot + 255) / 256, 256, 0, stream>>>(Wq, Wqp, DIM_, DIM_);
    pack_b_kernel<<<(tot + 255) / 256, 256, 0, stream>>>(Wo, Wop, DIM_, DIM_);
  }
  {
    int tot = (512 / 16) * (DIM_ / 32) * 32;
    pack_b_kernel<<<(tot + 255) / 256, 256, 0, stream>>>(Wk, Wkp, DIM_, 512);
    pack_b_kernel<<<(tot + 255) / 256, 256, 0, stream>>>(Wv, Wvp, DIM_, 512);
  }

  // 2) QKV projections (bias fused), block tile 128x128
  gemm_bf16_kernel<<<dim3(S_ / 128, DIM_ / 128), 256, 0, stream>>>(Xp, Wqp, bq, qraw, S_, DIM_, DIM_);
  gemm_bf16_kernel<<<dim3(S_ / 128, 512 / 128), 256, 0, stream>>>(Xp, Wkp, bk, kraw, S_, 512, DIM_);
  gemm_bf16_kernel<<<dim3(S_ / 128, 512 / 128), 256, 0, stream>>>(Xp, Wvp, bv, vraw, S_, 512, DIM_);

  // 3) RoPE + attention-friendly bf16 layouts
  rope_pack_q_kernel<<<(S_ * (DIM_ / 2) + 255) / 256, 256, 0, stream>>>(qraw, fcos, fsin, Qb);
  rope_pack_k_kernel<<<(S_ * 256 + 255) / 256, 256, 0, stream>>>(kraw, fcos, fsin, Kb);
  pack_v_kernel<<<(S_ * 512 + 255) / 256, 256, 0, stream>>>(vraw, Vt);

  // 4) flash attention: 32 heads * 128 qtiles = 4096 waves, 8 waves/block
  attn_kernel<<<(H_ * (S_ / 16)) / 8, 256, 0, stream>>>(Qb, Kb, Vt, attnO);

  // 5) output projection
  {
    int tot = (S_ / 16) * (DIM_ / 32) * 32;
    pack_a_kernel<<<(tot + 255) / 256, 256, 0, stream>>>(attnO, Ap2, S_, DIM_);
  }
  gemm_bf16_kernel<<<dim3(S_ / 128, DIM_ / 128), 256, 0, stream>>>(Ap2, Wop, bo, out, S_, DIM_, DIM_);
}
